// DeepseekV3Attention_33200097198621
// MI455X (gfx1250) — matrix operand order repaired and checked by reading, not compiled, not run on hardware
//
#include <hip/hip_runtime.h>

// ---------------------------------------------------------------------------
// Types for CDNA5 (wave32) WMMA + TDM
// ---------------------------------------------------------------------------
typedef _Float16 f16;
typedef __attribute__((ext_vector_type(16))) _Float16 v16h;
typedef __attribute__((ext_vector_type(8)))  _Float16 v8h;
typedef __attribute__((ext_vector_type(4)))  _Float16 v4h;
typedef __attribute__((ext_vector_type(8)))  float    v8f;
typedef __attribute__((ext_vector_type(4)))  float    v4f;
typedef __attribute__((ext_vector_type(4)))  unsigned int v4u;
typedef __attribute__((ext_vector_type(4)))  int      v4i;
typedef __attribute__((ext_vector_type(8)))  int      v8i;

#if defined(__gfx1250__) && __has_builtin(__builtin_amdgcn_tensor_load_to_lds) && \
    __has_builtin(__builtin_amdgcn_s_wait_tensorcnt)
#define USE_TDM 1
#else
#define USE_TDM 0
#endif

__device__ __forceinline__ v16h load16(const f16* lo, const f16* hi) {
    v16h r;
    ((v8h*)&r)[0] = *(const v8h*)lo;
    ((v8h*)&r)[1] = *(const v8h*)hi;
    return r;
}

__device__ __forceinline__ v8f wmma_f16(v16h a, v16h b, v8f c) {
    return __builtin_amdgcn_wmma_f32_16x16x32_f16(
        false, a, false, b, (short)0, c, false, false);
}

// ---------------------------------------------------------------------------
// Problem constants
// ---------------------------------------------------------------------------
#define SEQ   4096
#define HID   2048
#define NQH   16
#define NKVH  4
#define DH    128
#define QDIM  (NQH * DH)    // 2048
#define KVDIM (NKVH * DH)   // 512

// ---------------------------------------------------------------------------
// TDM: async DMA of a 128-row x 32-halfword row-major tile into LDS.
// D# layout per CDNA5 ISA ch.8 (group0 128b, group1 256b; groups2/3 zero
// for 2D tensors). data_size=1 (2 bytes). type=2 ("image").
// clang-22 (ROCm 7.2): 5-arg builtin; clang-23 (amdgpu-toolchain): 6-arg.
// ---------------------------------------------------------------------------
#if USE_TDM
__device__ __forceinline__ void tdm_issue(v4u g0, v8i g1) {
    v4i z4 = {0, 0, 0, 0};
#if __clang_major__ >= 23
    v8i z8 = {0, 0, 0, 0, 0, 0, 0, 0};
    __builtin_amdgcn_tensor_load_to_lds(g0, g1, z4, z4, z8, 0);
#else
    __builtin_amdgcn_tensor_load_to_lds(g0, g1, z4, z4, 0);
#endif
}

__device__ __forceinline__ void tdm_load_tile_128x32(const f16* gsrc,
                                                     unsigned lds_byte_off,
                                                     unsigned K, unsigned rowsRem) {
    unsigned long long ga = (unsigned long long)(size_t)gsrc;
    v4u g0 = { 1u,                                    // count=1 (valid, user mode)
               lds_byte_off,                          // lds_addr
               (unsigned)ga,                          // global_addr[31:0]
               (unsigned)(ga >> 32) | (2u << 30) };   // global_addr[56:32] | type=2
    unsigned long long st0 = (unsigned long long)K;   // tensor_dim0_stride (elements)
    v8i g1 = {
        (int)(1u << 16),                                           // data_size=1 (2B)
        (int)((K & 0xFFFFu) << 16),                                // tensor_dim0[15:0]
        (int)(((K >> 16) & 0xFFFFu) | ((rowsRem & 0xFFFFu) << 16)),// dim0 hi | dim1 lo
        (int)(((rowsRem >> 16) & 0xFFFFu) | (32u << 16)),          // dim1 hi | tile_dim0=32
        (int)128,                                                  // tile_dim1=128
        (int)(unsigned)(st0 & 0xFFFFFFFFu),                        // dim0_stride[31:0]
        (int)(unsigned)((st0 >> 32) & 0xFFFFu),                    // dim0_stride[47:32]
        0
    };
    tdm_issue(g0, g1);
}
#endif

// ---------------------------------------------------------------------------
// f32 -> f16 convert, 4-wide
// ---------------------------------------------------------------------------
__global__ __launch_bounds__(256) void k_f32_to_f16_v4(const float* __restrict__ in,
                                                       f16* __restrict__ out, int n4) {
    int i = blockIdx.x * 256 + threadIdx.x;
    if (i < n4) {
        v4f x = ((const v4f*)in)[i];
        v4h y;
#pragma unroll
        for (int e = 0; e < 4; ++e) y[e] = (f16)x[e];
        ((v4h*)out)[i] = y;
    }
}

// ---------------------------------------------------------------------------
// Transpose-convert: W (K x N, f32) -> Wt (N x K, f16), 32x32 LDS tiles
// ---------------------------------------------------------------------------
__global__ __launch_bounds__(256) void k_transpose_to_f16(const float* __restrict__ W,
                                                          f16* __restrict__ Wt,
                                                          int K, int N) {
    __shared__ float tile[32][33];
    const int n0 = blockIdx.x * 32, k0 = blockIdx.y * 32;
    const int tx = threadIdx.x, ty = threadIdx.y;   // (32, 8)
#pragma unroll
    for (int j = 0; j < 32; j += 8)
        tile[ty + j][tx] = W[(size_t)(k0 + ty + j) * N + (n0 + tx)];
    __syncthreads();
#pragma unroll
    for (int j = 0; j < 32; j += 8)
        Wt[(size_t)(n0 + ty + j) * K + (k0 + tx)] = (f16)tile[tx][ty + j];
}

// ---------------------------------------------------------------------------
// RoPE on (S, nheads, 128) f32 -> f16; position_ids == arange(S)
// ---------------------------------------------------------------------------
__global__ __launch_bounds__(256) void k_rope_to_f16(const float* __restrict__ X,
                                                     f16* __restrict__ Out,
                                                     int nheads, int total) {
    int idx = blockIdx.x * 256 + threadIdx.x;
    if (idx >= total) return;
    int j = idx & 63;
    int t = idx >> 6;
    int h = t % nheads;
    int s = t / nheads;
    size_t base = ((size_t)s * nheads + h) * DH;
    float x1 = X[base + j];
    float x2 = X[base + 64 + j];
    float freq = powf(10000.0f, -(float)j * (1.0f / 64.0f)) * (float)s;
    float c = cosf(freq), sn = sinf(freq);
    Out[base + j]      = (f16)(x1 * c - x2 * sn);
    Out[base + 64 + j] = (f16)(x2 * c + x1 * sn);
}

// ---------------------------------------------------------------------------
// Shared WMMA compute step for the GEMM: 8 waves, wave (wm,wn) owns 32x64.
// Abuf/Bbuf are 128x32 row-major f16 LDS tiles ((row,k) / (n,k)).
// ---------------------------------------------------------------------------
__device__ __forceinline__ void gemm_step(const f16* Abuf, const f16* Bbuf,
                                          int wm, int wn, int lh, int ln,
                                          v8f acc[2][4]) {
    v16h af[2], bf[4];
#pragma unroll
    for (int mi = 0; mi < 2; ++mi) {
        const f16* p = Abuf + (wm * 32 + mi * 16 + ln) * 32 + lh * 8;
        af[mi] = load16(p, p + 16);     // A frag: K halves {kg*8..+7, 16+kg*8..+7}
    }
#pragma unroll
    for (int ni = 0; ni < 4; ++ni) {
        const f16* p = Bbuf + (wn * 64 + ni * 16 + ln) * 32 + lh * 8;
        bf[ni] = load16(p, p + 16);     // B frag: K halves {kg*8..+7, 16+kg*8..+7}
    }
#pragma unroll
    for (int mi = 0; mi < 2; ++mi)
#pragma unroll
        for (int ni = 0; ni < 4; ++ni)
            acc[mi][ni] = wmma_f16(af[mi], bf[ni], acc[mi][ni]);
}

// ---------------------------------------------------------------------------
// WMMA GEMM: C(MxN,f32) = A(MxK,f16) * Bt(NxK,f16)^T   (Bt row-major N x K)
// Tile 128x128, K-step 32. TDM double-buffered async LDS loads when available;
// otherwise register-pipelined manual staging.
// ---------------------------------------------------------------------------
__global__ __launch_bounds__(256) void k_gemm_f16_wmma(const f16* __restrict__ A,
                                                       const f16* __restrict__ Bt,
                                                       float* __restrict__ C,
                                                       int M, int N, int K) {
    __shared__ f16 As[2][128 * 32];
    __shared__ f16 Bs[2][128 * 32];

    const int tid  = threadIdx.x;
    const int wave = tid >> 5, lane = tid & 31;
    const int wm = wave >> 1, wn = wave & 1;
    const int lh = lane >> 4, ln = lane & 15;
    const int m0 = blockIdx.y * 128;
    const int n0 = blockIdx.x * 128;

    const v8f vzero = {0.f,0.f,0.f,0.f,0.f,0.f,0.f,0.f};
    v8f acc[2][4];
#pragma unroll
    for (int mi = 0; mi < 2; ++mi)
#pragma unroll
        for (int ni = 0; ni < 4; ++ni) acc[mi][ni] = vzero;

#if USE_TDM
    const f16* aTile = A  + (size_t)m0 * K;
    const f16* bTile = Bt + (size_t)n0 * K;
    if (tid < 32) {   // wave 0 drives the Tensor Data Mover
        tdm_load_tile_128x32(aTile, (unsigned)(size_t)&As[0][0], (unsigned)K, (unsigned)(M - m0));
        tdm_load_tile_128x32(bTile, (unsigned)(size_t)&Bs[0][0], (unsigned)K, (unsigned)(N - n0));
    }
    int cur = 0;
    for (int k0 = 0; k0 < K; k0 += 32) {
        __syncthreads();                       // prev compute done -> other buffer free
        if (tid < 32) {
            if (k0 + 32 < K) {
                tdm_load_tile_128x32(aTile + k0 + 32, (unsigned)(size_t)&As[cur ^ 1][0],
                                     (unsigned)K, (unsigned)(M - m0));
                tdm_load_tile_128x32(bTile + k0 + 32, (unsigned)(size_t)&Bs[cur ^ 1][0],
                                     (unsigned)K, (unsigned)(N - n0));
                __builtin_amdgcn_s_wait_tensorcnt(2);   // current tile's 2 ops retired
            } else {
                __builtin_amdgcn_s_wait_tensorcnt(0);
            }
        }
        __syncthreads();                       // buf[cur] visible to all waves
        gemm_step(&As[cur][0], &Bs[cur][0], wm, wn, lh, ln, acc);
        cur ^= 1;
    }
#else
    // Register-pipelined manual staging: global loads for tile k+1 are issued
    // right after the barrier and consumed one iteration later, so the
    // s_wait_loadcnt is hidden behind a full tile of WMMA compute.
    const int srow = tid >> 1, sc = (tid & 1) * 16;
    const f16* pA = A  + (size_t)(m0 + srow) * K + sc;
    const f16* pB = Bt + (size_t)(n0 + srow) * K + sc;
    f16* sA = &As[0][srow * 32 + sc];
    f16* sB = &Bs[0][srow * 32 + sc];

    v8h ra0 = *(const v8h*)(pA);
    v8h ra1 = *(const v8h*)(pA + 8);
    v8h rb0 = *(const v8h*)(pB);
    v8h rb1 = *(const v8h*)(pB + 8);
    for (int k0 = 0; k0 < K; k0 += 32) {
        __syncthreads();                       // prev compute done -> LDS free
        *(v8h*)sA       = ra0;
        *(v8h*)(sA + 8) = ra1;
        *(v8h*)sB       = rb0;
        *(v8h*)(sB + 8) = rb1;
        __syncthreads();                       // tile visible to all waves
        if (k0 + 32 < K) {
            ra0 = *(const v8h*)(pA + k0 + 32);
            ra1 = *(const v8h*)(pA + k0 + 40);
            rb0 = *(const v8h*)(pB + k0 + 32);
            rb1 = *(const v8h*)(pB + k0 + 40);
            __builtin_prefetch(pA + k0 + 64, 0, 1);
            __builtin_prefetch(pB + k0 + 64, 0, 1);
        }
        gemm_step(&As[0][0], &Bs[0][0], wm, wn, lh, ln, acc);
    }
#endif

    // Epilogue: C/D layout -> VGPR r = row r (lanes 0-15) / r+8 (lanes 16-31)
#pragma unroll
    for (int mi = 0; mi < 2; ++mi)
#pragma unroll
        for (int ni = 0; ni < 4; ++ni) {
            float* cp = C + (size_t)(m0 + wm * 32 + mi * 16 + 8 * lh) * N
                          + (n0 + wn * 64 + ni * 16 + ln);
#pragma unroll
            for (int r = 0; r < 8; ++r)
                cp[(size_t)r * N] = acc[mi][ni][r];
        }
}

// ---------------------------------------------------------------------------
// Flash attention (causal, GQA) with WMMA. grid=(S/64, NQH), block=128.
// Q (S,16,128) f16; K,V (S,4,128) f16; output Oh (S,2048) f16.
// ---------------------------------------------------------------------------
__global__ __launch_bounds__(128) void k_attn_wmma(const f16* __restrict__ Qh,
                                                   const f16* __restrict__ Kh,
                                                   const f16* __restrict__ Vh,
                                                   f16* __restrict__ Oh) {
    const float scale = 0.08838834764831845f;   // 1/sqrt(128)
    __shared__ f16 Vt[DH * 32];                 // V tile transposed (d, key)
    __shared__ f16 Pbuf[4][16 * 32];            // per-wave P relayout

    const int tid  = threadIdx.x;
    const int wave = tid >> 5, lane = tid & 31;
    const int lh = lane >> 4, ln = lane & 15;
    const int qb    = blockIdx.x * 64;
    const int h     = blockIdx.y;
    const int kvh   = h >> 2;
    const int qbase = qb + wave * 16;

    const v8f vzero = {0.f,0.f,0.f,0.f,0.f,0.f,0.f,0.f};
    v8f acc[8];
    float m[8], l[8];
#pragma unroll
    for (int i = 0; i < 8; ++i) { acc[i] = vzero; m[i] = -3.0e38f; l[i] = 0.f; }

    v16h qf[4];
    {
        const f16* qrow = Qh + ((size_t)(qbase + ln) * NQH + h) * DH;
#pragma unroll
        for (int c = 0; c < 4; ++c) {
            const f16* p = qrow + c * 32 + lh * 8;
            qf[c] = load16(p, p + 16);
        }
    }

    const int ktend = qb + 64;
    for (int kt = 0; kt < ktend; kt += 32) {
        __syncthreads();
        // Stage V(32 keys x 128 d) -> Vt(d, key); b128 global reads
        for (int i = tid; i < 32 * 16; i += 128) {
            int key = i >> 4, c8 = (i & 15) * 8;
            v8h v = *(const v8h*)(Vh + ((size_t)(kt + key) * NKVH + kvh) * DH + c8);
#pragma unroll
            for (int e = 0; e < 8; ++e) Vt[(c8 + e) * 32 + key] = v[e];
        }
        __syncthreads();

        if (kt <= qbase + 15) {              // wave-uniform causal skip
            v8f s0 = vzero, s1 = vzero;
#pragma unroll
            for (int c = 0; c < 4; ++c) {
                const f16* kp0 = Kh + ((size_t)(kt + ln) * NKVH + kvh) * DH + c * 32 + lh * 8;
                s0 = wmma_f16(qf[c], load16(kp0, kp0 + 16), s0);
                const f16* kp1 = Kh + ((size_t)(kt + 16 + ln) * NKVH + kvh) * DH + c * 32 + lh * 8;
                s1 = wmma_f16(qf[c], load16(kp1, kp1 + 16), s1);
            }

            float fsc[8];
#pragma unroll
            for (int r = 0; r < 8; ++r) {
                int qg = qbase + r + 8 * lh;
                float a0 = (kt + ln      <= qg) ? s0[r] * scale : -1.0e30f;
                float a1 = (kt + 16 + ln <= qg) ? s1[r] * scale : -1.0e30f;
                float mx = fmaxf(a0, a1);
                mx = fmaxf(mx, __shfl_xor(mx, 1));
                mx = fmaxf(mx, __shfl_xor(mx, 2));
                mx = fmaxf(mx, __shfl_xor(mx, 4));
                mx = fmaxf(mx, __shfl_xor(mx, 8));
                float mn = fmaxf(m[r], mx);
                float f  = __expf(m[r] - mn);
                m[r] = mn;
                float p0 = __expf(a0 - mn);
                float p1 = __expf(a1 - mn);
                float rs = p0 + p1;
                rs += __shfl_xor(rs, 1);
                rs += __shfl_xor(rs, 2);
                rs += __shfl_xor(rs, 4);
                rs += __shfl_xor(rs, 8);
                l[r] = l[r] * f + rs;
                fsc[r] = f;
                int row = r + 8 * lh;
                Pbuf[wave][row * 32 + ln]      = (f16)p0;
                Pbuf[wave][row * 32 + 16 + ln] = (f16)p1;
            }
#pragma unroll
            for (int dt = 0; dt < 8; ++dt)
#pragma unroll
                for (int r = 0; r < 8; ++r)
                    acc[dt][r] *= fsc[r];

            const f16* pp = &Pbuf[wave][ln * 32 + lh * 8];
            v16h pa = load16(pp, pp + 16);
#pragma unroll
            for (int dt = 0; dt < 8; ++dt) {
                const f16* vp = &Vt[(dt * 16 + ln) * 32 + lh * 8];
                acc[dt] = wmma_f16(pa, load16(vp, vp + 16), acc[dt]);
            }
        }
    }

    float inv[8];
#pragma unroll
    for (int r = 0; r < 8; ++r) inv[r] = 1.0f / l[r];
#pragma unroll
    for (int dt = 0; dt < 8; ++dt)
#pragma unroll
        for (int r = 0; r < 8; ++r) {
            int row = qbase + r + 8 * lh;
            int col = h * DH + dt * 16 + ln;
            Oh[(size_t)row * QDIM + col] = (f16)(acc[dt][r] * inv[r]);
        }
}

// ---------------------------------------------------------------------------
// Host-side launch
// ---------------------------------------------------------------------------
static inline size_t alignup(size_t x) { return (x + 255) & ~(size_t)255; }

extern "C" void kernel_launch(void* const* d_in, const int* in_sizes, int n_in,
                              void* d_out, int out_size, void* d_ws, size_t ws_size,
                              hipStream_t stream) {
    (void)in_sizes; (void)n_in; (void)out_size; (void)ws_size;
    const float* x  = (const float*)d_in[0];   // (S, H)
    const float* Wq = (const float*)d_in[1];   // (H, 2048)
    const float* Wk = (const float*)d_in[2];   // (H, 512)
    const float* Wv = (const float*)d_in[3];   // (H, 512)
    const float* Wo = (const float*)d_in[4];   // (2048, H)
    float* out = (float*)d_out;

    char* ws = (char*)d_ws;
    size_t off = 0;
    f16* xh   = (f16*)(ws + off); off = alignup(off + (size_t)SEQ * HID * 2);
    f16* Wqt  = (f16*)(ws + off); off = alignup(off + (size_t)QDIM * HID * 2);   // (2048, H)
    f16* Wkt  = (f16*)(ws + off); off = alignup(off + (size_t)KVDIM * HID * 2);  // (512, H)
    f16* Wvt  = (f16*)(ws + off); off = alignup(off + (size_t)KVDIM * HID * 2);
    f16* Wot  = (f16*)(ws + off); off = alignup(off + (size_t)HID * QDIM * 2);   // (H, 2048)
    float* Qf = (float*)(ws + off); off = alignup(off + (size_t)SEQ * QDIM * 4);
    float* Kf = (float*)(ws + off); off = alignup(off + (size_t)SEQ * KVDIM * 4);
    float* Vf = (float*)(ws + off); off = alignup(off + (size_t)SEQ * KVDIM * 4);
    f16* Qh2  = (f16*)(ws + off);   off = alignup(off + (size_t)SEQ * QDIM * 2);
    f16* Kh2  = (f16*)(ws + off);   off = alignup(off + (size_t)SEQ * KVDIM * 2);
    f16* Vh2  = (f16*)(ws + off);   off = alignup(off + (size_t)SEQ * KVDIM * 2);
    f16* Oh   = (f16*)(ws + off);   off = alignup(off + (size_t)SEQ * QDIM * 2);

    // 1) activations f32->f16; weights transposed-converted to (N, K) f16
    {
        int n4 = SEQ * HID / 4;
        k_f32_to_f16_v4<<<(n4 + 255) / 256, 256, 0, stream>>>(x, xh, n4);
        dim3 blk(32, 8);
        k_transpose_to_f16<<<dim3(QDIM / 32, HID / 32), blk, 0, stream>>>(Wq, Wqt, HID, QDIM);
        k_transpose_to_f16<<<dim3(KVDIM / 32, HID / 32), blk, 0, stream>>>(Wk, Wkt, HID, KVDIM);
        k_transpose_to_f16<<<dim3(KVDIM / 32, HID / 32), blk, 0, stream>>>(Wv, Wvt, HID, KVDIM);
        k_transpose_to_f16<<<dim3(HID / 32, QDIM / 32), blk, 0, stream>>>(Wo, Wot, QDIM, HID);
    }

    // 2) QKV projections (WMMA GEMMs, TDM-fed LDS tiles when available)
    k_gemm_f16_wmma<<<dim3(QDIM / 128, SEQ / 128), 256, 0, stream>>>(xh, Wqt, Qf, SEQ, QDIM, HID);
    k_gemm_f16_wmma<<<dim3(KVDIM / 128, SEQ / 128), 256, 0, stream>>>(xh, Wkt, Kf, SEQ, KVDIM, HID);
    k_gemm_f16_wmma<<<dim3(KVDIM / 128, SEQ / 128), 256, 0, stream>>>(xh, Wvt, Vf, SEQ, KVDIM, HID);

    // 3) RoPE(Q), RoPE(K) fused with f32->f16; plain convert for V
    {
        int nq = SEQ * NQH * 64;
        k_rope_to_f16<<<(nq + 255) / 256, 256, 0, stream>>>(Qf, Qh2, NQH, nq);
        int nk = SEQ * NKVH * 64;
        k_rope_to_f16<<<(nk + 255) / 256, 256, 0, stream>>>(Kf, Kh2, NKVH, nk);
        int n4 = SEQ * KVDIM / 4;
        k_f32_to_f16_v4<<<(n4 + 255) / 256, 256, 0, stream>>>(Vf, Vh2, n4);
    }

    // 4) causal GQA flash attention with WMMA, f16 output
    k_attn_wmma<<<dim3(SEQ / 64, NQH), 128, 0, stream>>>(Qh2, Kh2, Vh2, Oh);

    // 5) output projection
    k_gemm_f16_wmma<<<dim3(HID / 128, SEQ / 128), 256, 0, stream>>>(Oh, Wot, out, SEQ, HID, QDIM);
}
